// HyperNetwork_FC_VQVAE_68874095558707
// MI455X (gfx1250) — matrix operand order repaired and checked by reading, not compiled, not run on hardware
//
#include <hip/hip_runtime.h>
#include <hip/hip_bf16.h>

// ---------------------------------------------------------------------------
// HyperNetwork_FC_VQVAE on MI455X (gfx1250, wave32, WMMA)
//
//   K0 prep:      embd row norms + zero loss accumulators
//   K1 enc+VQ:    encoder MLP, nearest-code argmin (codebook in LDS)
//   K2 dec+proj:  decoder MLP, recon loss, fused GRU input projection
//   K3 gru+mlp:   31-step GRU scan (1 thread / batch row) + trigger MLP -> u
//   K4 gemm x2:   hypernetwork layers as bf16 WMMA GEMMs; A = outer(u,x)
//                 built on the fly in LDS; bias-augmented K; scalar-uniform
//                 fast path for main-weight chunks (branch-free fill,
//                 native V_CVT bf16 packing)
//   K5 finalize:  combine loss sums -> d_out tail
// ---------------------------------------------------------------------------

#define B_   2048
#define S_   31
#define D_   32
#define K_   450
#define HVQ_ 64
#define ZD_  32
#define U1_  256
#define U2_  128
#define NR_  (B_ * S_)          // 63488 rows = 248 * 256

typedef __attribute__((ext_vector_type(16))) __bf16 v16bf;
typedef __attribute__((ext_vector_type(8)))  float  v8f;

union Frag16 { v16bf v; unsigned int u[8]; };
union Acc8   { v8f  v; float f[8]; };

// native float->bf16 conversion (lowers to V_CVT_*BF16* on gfx1250, RNE)
static __device__ __forceinline__ unsigned short f2bf(float x) {
  __bf16 b = (__bf16)x;
  return __builtin_bit_cast(unsigned short, b);
}
static __device__ __forceinline__ unsigned int pack2bf(float a, float b) {
  return (unsigned int)f2bf(a) | ((unsigned int)f2bf(b) << 16);
}
static __device__ __forceinline__ float sigm(float x) {
  return 1.0f / (1.0f + __expf(-x));
}

// ---------------------------------------------------------------- K0: prep
__global__ void prep_kernel(const float* __restrict__ embd,
                            float* __restrict__ enorm,
                            float* __restrict__ lossAcc) {
  int t = threadIdx.x;
  if (t < 3) lossAcc[t] = 0.0f;
  for (int k = t; k < K_; k += blockDim.x) {
    float s = 0.0f;
    for (int j = 0; j < ZD_; ++j) { float e = embd[k * ZD_ + j]; s += e * e; }
    enorm[k] = s;
  }
}

// ------------------------------------------------------ K1: encoder + VQ
__global__ __launch_bounds__(256) void encvq_kernel(
    const float* __restrict__ z,
    const float* __restrict__ ew1, const float* __restrict__ eb1,
    const float* __restrict__ ew2, const float* __restrict__ eb2,
    const float* __restrict__ embd, const float* __restrict__ enorm,
    int* __restrict__ idxOut, float* __restrict__ lossAcc) {
  __shared__ float sEmb[K_ * ZD_];     // 57.6 KB codebook in LDS
  __shared__ float sNorm[K_];
  __shared__ float sRed[256];
  int t = threadIdx.x;
  for (int i = t; i < K_ * ZD_; i += 256) sEmb[i]  = embd[i];
  for (int i = t; i < K_;       i += 256) sNorm[i] = enorm[i];
  __syncthreads();

  int rid = blockIdx.x * 256 + t;      // exactly covers NR_
  float zr[D_], ze[ZD_];
  for (int j = 0; j < D_;  ++j) zr[j] = z[rid * D_ + j];
  for (int o = 0; o < ZD_; ++o) ze[o] = eb2[o];
  for (int hh = 0; hh < HVQ_; ++hh) {
    float hv = eb1[hh];
    for (int j = 0; j < D_; ++j) hv += zr[j] * ew1[j * HVQ_ + hh];
    hv = hv > 0.0f ? hv : 0.0f;
    for (int o = 0; o < ZD_; ++o) ze[o] += hv * ew2[hh * ZD_ + o];
  }
  float zn = 0.0f;
  for (int o = 0; o < ZD_; ++o) zn += ze[o] * ze[o];
  float best = 3.4e38f; int bi = 0;
  for (int k = 0; k < K_; ++k) {
    const float* e = &sEmb[k * ZD_];
    float dp = 0.0f;
    for (int o = 0; o < ZD_; ++o) dp += ze[o] * e[o];
    float d = zn - 2.0f * dp + sNorm[k];
    if (d < best) { best = d; bi = k; }
  }
  idxOut[rid] = bi;
  float cm = 0.0f, zem = 0.0f;
  const float* e = &sEmb[bi * ZD_];
  for (int o = 0; o < ZD_; ++o) {
    float d1 = ze[o] - e[o]; cm  += d1 * d1;
    float d2 = zr[o] - e[o]; zem += d2 * d2;
  }
  sRed[t] = cm; __syncthreads();
  for (int st = 128; st > 0; st >>= 1) { if (t < st) sRed[t] += sRed[t + st]; __syncthreads(); }
  if (t == 0) atomicAdd(&lossAcc[1], sRed[0]);
  __syncthreads();
  sRed[t] = zem; __syncthreads();
  for (int st = 128; st > 0; st >>= 1) { if (t < st) sRed[t] += sRed[t + st]; __syncthreads(); }
  if (t == 0) atomicAdd(&lossAcc[2], sRed[0]);
}

// ---------------------------------- K2: decoder + recon loss + GRU xproj
__global__ __launch_bounds__(256) void decproj_kernel(
    const float* __restrict__ z, const float* __restrict__ embd,
    const int* __restrict__ idxIn,
    const float* __restrict__ dw1, const float* __restrict__ db1,
    const float* __restrict__ dw2, const float* __restrict__ db2,
    const float* __restrict__ wih, const float* __restrict__ bih,
    float* __restrict__ xproj, float* __restrict__ lossAcc) {
  __shared__ float sDw1[ZD_ * HVQ_];
  __shared__ float sDw2[HVQ_ * D_];
  __shared__ float sWih[96 * D_];
  __shared__ float sB[HVQ_ + D_ + 96];
  __shared__ float sRed[256];
  int t = threadIdx.x;
  for (int i = t; i < ZD_ * HVQ_; i += 256) sDw1[i] = dw1[i];
  for (int i = t; i < HVQ_ * D_;  i += 256) sDw2[i] = dw2[i];
  for (int i = t; i < 96 * D_;    i += 256) sWih[i] = wih[i];
  if (t < HVQ_) sB[t] = db1[t];
  if (t < D_)   sB[HVQ_ + t] = db2[t];
  if (t < 96)   sB[HVQ_ + D_ + t] = bih[t];
  __syncthreads();

  int rid = blockIdx.x * 256 + t;
  int bi = idxIn[rid];
  float ze[ZD_], rec[D_];
  for (int o = 0; o < ZD_; ++o) ze[o]  = embd[bi * ZD_ + o];
  for (int o = 0; o < D_;  ++o) rec[o] = sB[HVQ_ + o];
  for (int hh = 0; hh < HVQ_; ++hh) {
    float hv = sB[hh];
    for (int j = 0; j < ZD_; ++j) hv += ze[j] * sDw1[j * HVQ_ + hh];
    hv = hv > 0.0f ? hv : 0.1f * hv;                // leaky relu
    for (int o = 0; o < D_; ++o) rec[o] += hv * sDw2[hh * D_ + o];
  }
  float rl = 0.0f;
  for (int o = 0; o < D_; ++o) { float d = rec[o] - z[rid * D_ + o]; rl += d * d; }
  for (int o = 0; o < 96; ++o) {
    float s = sB[HVQ_ + D_ + o];
    for (int j = 0; j < D_; ++j) s += rec[j] * sWih[o * D_ + j];
    xproj[rid * 96 + o] = s;
  }
  sRed[t] = rl; __syncthreads();
  for (int st = 128; st > 0; st >>= 1) { if (t < st) sRed[t] += sRed[t + st]; __syncthreads(); }
  if (t == 0) atomicAdd(&lossAcc[0], sRed[0]);
}

// ------------------------------------------------- K3: GRU scan + MLP -> u
__global__ __launch_bounds__(256) void gru_kernel(
    const float* __restrict__ xproj,
    const float* __restrict__ whh, const float* __restrict__ bhh,
    const float* __restrict__ mw1, const float* __restrict__ mb1,
    const float* __restrict__ mw2, const float* __restrict__ mb2,
    float* __restrict__ uOut) {
  __shared__ float sWhh[96 * D_];
  __shared__ float sBhh[96];
  __shared__ float sMw1[D_ * D_], sMw2[D_ * D_];
  __shared__ float sMb1[D_], sMb2[D_];
  int t = threadIdx.x;
  for (int i = t; i < 96 * D_; i += 256) sWhh[i] = whh[i];
  if (t < 96) sBhh[t] = bhh[t];
  for (int i = t; i < D_ * D_; i += 256) { sMw1[i] = mw1[i]; sMw2[i] = mw2[i]; }
  if (t < D_) { sMb1[t] = mb1[t]; sMb2[t] = mb2[t]; }
  __syncthreads();

  int b = blockIdx.x * 256 + t;
  float h[D_];
  for (int j = 0; j < D_; ++j) h[j] = 0.0f;
  for (int s = 0; s < S_; ++s) {
    const float* xt = &xproj[(b * S_ + s) * 96];
    float hn[D_];
    for (int j = 0; j < D_; ++j) {
      float hr = sBhh[j], hu = sBhh[32 + j], hv = sBhh[64 + j];
      const float* wr = &sWhh[j * D_];
      const float* wu = &sWhh[(32 + j) * D_];
      const float* wn = &sWhh[(64 + j) * D_];
      for (int d = 0; d < D_; ++d) {
        float hd = h[d]; hr += hd * wr[d]; hu += hd * wu[d]; hv += hd * wn[d];
      }
      float r  = sigm(xt[j]      + hr);
      float uu = sigm(xt[32 + j] + hu);
      float nn = tanhf(xt[64 + j] + r * hv);
      hn[j] = (1.0f - uu) * nn + uu * h[j];
    }
    for (int j = 0; j < D_; ++j) h[j] = hn[j];
  }
  float tt[D_];
  for (int o = 0; o < D_; ++o) {
    float s2 = sMb1[o];
    for (int d = 0; d < D_; ++d) s2 += h[d] * sMw1[d * D_ + o];
    tt[o] = tanhf(s2);
  }
  for (int o = 0; o < D_; ++o) {
    float s2 = sMb2[o];
    for (int d = 0; d < D_; ++d) s2 += tt[d] * sMw2[d * D_ + o];
    uOut[b * D_ + o] = s2;
  }
}

// ---------------------- K4: hypernetwork layer as bf16 WMMA GEMM --------
// y[b,n] = act( sum_k P[b,k] * W[k,n] ), with augmented K:
//   k in [0,Kmain)          : P = u[b,d]*xv[b,i]   W = hw1 (Kmain x N)
//   k in [Kmain,Kmain+iDim) : P = xv[b,i]          W = hb1 (iDim x N)
//   k in [.., +32)          : P = u[b,d]           W = hw2 (32 x N)
//   k == Kone               : P = 1                W = hb2 (N)
//   k > Kone (pad to 32)    : 0
// Kmain is a multiple of KSTEP, and KSTEP | iDim, so every k-chunk is either
// wholly inside the main region (uniform d => branch-free outer-product fill)
// or wholly in the bias tail (rare; generic predicated fill).
#define TM 128
#define TN 128
#define KSTEP 32
#define AST 34    // LDS k-stride (bf16 units); AST/2 = 17 dwords per row

__global__ __launch_bounds__(256) void hyper_gemm_kernel(
    const float* __restrict__ u,   const float* __restrict__ xv,
    const float* __restrict__ w,   const float* __restrict__ b1w,
    const float* __restrict__ w2,  const float* __restrict__ b2,
    float* __restrict__ out,
    int iDim, int iShift, int Kmain, int Ktot, int N, int applyTanh) {
  __shared__ __align__(16) unsigned short Al[TM * AST];  // A tile [row][k]
  __shared__ __align__(16) unsigned short Bl[TN * AST];  // B tile [n][k]
  __shared__ float uS[TM * 32];                          // block's u rows
  int t = threadIdx.x;
  int lane = t & 31, wave = t >> 5;
  int waveM = wave & 3, waveN = wave >> 2;  // 4x2 wave grid over 128x128
  int m0 = blockIdx.x * TM;
  int n0 = blockIdx.y * TN;
  int Kbias1 = Kmain + iDim;
  int Kone   = Kmain + iDim + 32;
  unsigned int* Alw = (unsigned int*)Al;
  unsigned int* Blw = (unsigned int*)Bl;

  for (int i = t; i < TM * 32; i += 256)
    uS[i] = u[(m0 + (i >> 5)) * 32 + (i & 31)];

  v8f acc[2][4];
  for (int f = 0; f < 2; ++f)
    for (int g = 0; g < 4; ++g) acc[f][g] = (v8f)(0.0f);

  __syncthreads();

  int ksteps = Ktot / KSTEP;
  for (int ks = 0; ks < ksteps; ++ks) {
    int k0 = ks * KSTEP;
    if (k0 + KSTEP <= Kmain) {
      // ---- fast path: whole chunk in main region; d uniform ----
      int d  = k0 >> iShift;
      int i0 = k0 & (iDim - 1);
      // A tile: outer product uS[:,d] x xv[:, i0..i0+31]; dword-packed
      for (int e = t; e < TM * 16; e += 256) {
        int r = e >> 4, kp = e & 15;
        int row = m0 + r;
        float uv = uS[r * 32 + d];
        float x0 = xv[row * iDim + i0 + 2 * kp];
        float x1 = xv[row * iDim + i0 + 2 * kp + 1];
        Alw[r * 17 + kp] = pack2bf(uv * x0, uv * x1);
      }
      // B tile: straight weight rows, stored transposed [n][k]
      for (int e = t; e < TN * 16; e += 256) {
        int n = e & (TN - 1), kp = e >> 7;
        int gn = n0 + n;
        float w0 = w[(k0 + 2 * kp) * N + gn];
        float w1 = w[(k0 + 2 * kp + 1) * N + gn];
        Blw[n * 17 + kp] = pack2bf(w0, w1);
      }
      // prefetch next chunk of the weight stream (global_prefetch_b8)
      if (t < TN && k0 + KSTEP < Kmain)
        __builtin_prefetch(&w[(k0 + KSTEP) * N + n0 + t], 0, 1);
    } else {
      // ---- generic tail: bias-augmented rows (rare) ----
      for (int e = t; e < TM * KSTEP; e += 256) {
        int r = e >> 5, kk = e & 31;
        int gk = k0 + kk, row = m0 + r;
        float val;
        if (gk < Kmain) {
          int dd = gk >> iShift, ii = gk & (iDim - 1);
          val = uS[r * 32 + dd] * xv[row * iDim + ii];
        } else if (gk < Kbias1) val = xv[row * iDim + (gk - Kmain)];
        else if (gk < Kone)     val = uS[r * 32 + (gk - Kbias1)];
        else if (gk == Kone)    val = 1.0f;
        else                    val = 0.0f;
        Al[r * AST + kk] = f2bf(val);
      }
      for (int e = t; e < TN * KSTEP; e += 256) {
        int n = e & (TN - 1), kk = e >> 7;
        int gk = k0 + kk, gn = n0 + n;
        float val;
        if (gk < Kmain)         val = w[gk * N + gn];
        else if (gk < Kbias1)   val = b1w[(gk - Kmain) * N + gn];
        else if (gk < Kone)     val = w2[(gk - Kbias1) * N + gn];
        else if (gk == Kone)    val = b2[gn];
        else                    val = 0.0f;
        Bl[n * AST + kk] = f2bf(val);
      }
    }
    __syncthreads();

    int khalf = lane >> 4, l16 = lane & 15;
    // B fragments: lane=col, VGPR v holds K=2v,2v+1 (+16 for upper lanes)
    Frag16 bf[4];
    for (int g = 0; g < 4; ++g) {
      int c = waveN * 64 + g * 16 + l16;
      const unsigned int* bp = (const unsigned int*)&Bl[c * AST];
      for (int v = 0; v < 8; ++v) {
        int kb = 2 * v + 16 * khalf;
        bf[g].u[v] = bp[kb >> 1];
      }
    }
    // A fragments: lane=row, K 0-15 (lower) vs 16-31 (upper)
    for (int f = 0; f < 2; ++f) {
      int r = waveM * 32 + f * 16 + l16;
      const unsigned int* ap = (const unsigned int*)&Al[r * AST];
      Frag16 af;
      for (int v = 0; v < 8; ++v) {
        int kb = 2 * v + 16 * khalf;
        af.u[v] = ap[kb >> 1];
      }
      for (int g = 0; g < 4; ++g) {
        acc[f][g] = __builtin_amdgcn_wmma_f32_16x16x32_bf16(
            false, af.v, false, bf[g].v, (short)0, acc[f][g], false, false);
      }
    }
    __syncthreads();
  }

  // store C/D: VGPR v -> row v (+8 for upper lanes), col = lane&15
  int khalf = lane >> 4, l16 = lane & 15;
  for (int f = 0; f < 2; ++f) {
    for (int g = 0; g < 4; ++g) {
      Acc8 a; a.v = acc[f][g];
      int col = n0 + waveN * 64 + g * 16 + l16;
      for (int v = 0; v < 8; ++v) {
        int row = m0 + waveM * 32 + f * 16 + v + 8 * khalf;
        float cv = a.f[v];
        if (applyTanh) cv = tanhf(cv);
        out[row * N + col] = cv;
      }
    }
  }
}

// --------------------------------------------------------- K5: finalize
__global__ void finalize_kernel(const float* __restrict__ lossAcc,
                                float* __restrict__ outLoss) {
  float nInv = 1.0f / ((float)NR_ * (float)D_);
  outLoss[0] = lossAcc[0] * nInv + 0.25f * (lossAcc[1] + lossAcc[2]) * nInv;
}

// ---------------------------------------------------------------------------
extern "C" void kernel_launch(void* const* d_in, const int* in_sizes, int n_in,
                              void* d_out, int out_size, void* d_ws, size_t ws_size,
                              hipStream_t stream) {
  (void)in_sizes; (void)n_in; (void)out_size; (void)ws_size;
  const float* x     = (const float*)d_in[0];
  const float* z     = (const float*)d_in[1];
  const float* ew1   = (const float*)d_in[2];
  const float* eb1   = (const float*)d_in[3];
  const float* ew2   = (const float*)d_in[4];
  const float* eb2   = (const float*)d_in[5];
  const float* embd  = (const float*)d_in[6];
  const float* dw1   = (const float*)d_in[7];
  const float* db1   = (const float*)d_in[8];
  const float* dw2   = (const float*)d_in[9];
  const float* db2   = (const float*)d_in[10];
  const float* wih   = (const float*)d_in[11];
  const float* whh   = (const float*)d_in[12];
  const float* bih   = (const float*)d_in[13];
  const float* bhh   = (const float*)d_in[14];
  const float* mw1   = (const float*)d_in[15];
  const float* mb1   = (const float*)d_in[16];
  const float* mw2   = (const float*)d_in[17];
  const float* mb2   = (const float*)d_in[18];
  const float* hw1_0 = (const float*)d_in[19];
  const float* hb1_0 = (const float*)d_in[20];
  const float* hw2_0 = (const float*)d_in[21];
  const float* hb2_0 = (const float*)d_in[22];
  const float* hw1_1 = (const float*)d_in[23];
  const float* hb1_1 = (const float*)d_in[24];
  const float* hw2_1 = (const float*)d_in[25];
  const float* hb2_1 = (const float*)d_in[26];

  // workspace layout (256-aligned)
  char* ws = (char*)d_ws;
  float* lossAcc = (float*)(ws + 0);                         // 3 floats
  float* enorm   = (float*)(ws + 256);                       // 450 floats
  int*   idx     = (int*)  (ws + 4096);                      // 63488 ints
  float* xproj   = (float*)(ws + 258048);                    // 63488*96 floats
  float* uBuf    = (float*)(ws + 258048 + (size_t)NR_*96*4); // 2048*32 floats
  float* y0      = (float*)((char*)uBuf + (size_t)B_*32*4);  // 2048*256 floats
  float* outX    = (float*)d_out;                            // 2048*128
  float* outLoss = outX + (size_t)B_ * U2_;                  // scalar

  prep_kernel<<<dim3(1), dim3(512), 0, stream>>>(embd, enorm, lossAcc);
  encvq_kernel<<<dim3(NR_ / 256), dim3(256), 0, stream>>>(
      z, ew1, eb1, ew2, eb2, embd, enorm, idx, lossAcc);
  decproj_kernel<<<dim3(NR_ / 256), dim3(256), 0, stream>>>(
      z, embd, idx, dw1, db1, dw2, db2, wih, bih, xproj, lossAcc);
  gru_kernel<<<dim3(B_ / 256), dim3(256), 0, stream>>>(
      xproj, whh, bhh, mw1, mb1, mw2, mb2, uBuf);
  // layer 0: Kmain=32*32=1024, iDim=32, N=256, Ktot=1120 (35 k-steps), tanh
  hyper_gemm_kernel<<<dim3(B_ / TM, U1_ / TN), dim3(256), 0, stream>>>(
      uBuf, x, hw1_0, hb1_0, hw2_0, hb2_0, y0, 32, 5, 1024, 1120, U1_, 1);
  // layer 1: Kmain=32*256=8192, iDim=256, N=128, Ktot=8512 (266 k-steps)
  hyper_gemm_kernel<<<dim3(B_ / TM, U2_ / TN), dim3(256), 0, stream>>>(
      uBuf, y0, hw1_1, hb1_1, hw2_1, hb2_1, outX, 256, 8, 8192, 8512, U2_, 0);
  finalize_kernel<<<dim3(1), dim3(1), 0, stream>>>(lossAcc, outLoss);
}
